// DiffAttention_7430293422338
// MI455X (gfx1250) — matrix operand order repaired and checked by reading, not compiled, not run on hardware
//
#include <hip/hip_runtime.h>
#include <hip/hip_bf16.h>

typedef __attribute__((ext_vector_type(16))) _Float16 v16h;
typedef __attribute__((ext_vector_type(8)))  _Float16 v8h;
typedef __attribute__((ext_vector_type(8)))  float    v8f;

// ---------------------------------------------------------------------------
// Async global->LDS copy via CDNA5 GLOBAL_LOAD_ASYNC_TO_LDS_B128 (ASYNCcnt).
// Inline asm (ISA 15.18.3 / CDNA5 ISA): VDST = per-lane LDS byte
// offset (generic __shared__ pointer truncates to LDS offset, ISA 10.2),
// VADDR = 64-bit global address.
// ---------------------------------------------------------------------------
__device__ __forceinline__ void cp16_g2s(void* lds, const void* g) {
    unsigned int laddr = (unsigned int)(unsigned long long)lds;
    asm volatile("global_load_async_to_lds_b128 %0, %1, off"
                 :
                 : "v"(laddr), "v"(g)
                 : "memory");
}
__device__ __forceinline__ void wait_cp() {
    asm volatile("s_wait_asynccnt 0" ::: "memory");
}

// Build a v16h fragment from two contiguous 16B chunks.
__device__ __forceinline__ v16h frag2(const _Float16* lo, const _Float16* hi) {
    v8h a = *reinterpret_cast<const v8h*>(lo);
    v8h b = *reinterpret_cast<const v8h*>(hi);
    return __builtin_shufflevector(a, b, 0, 1, 2, 3, 4, 5, 6, 7, 8, 9, 10, 11, 12, 13, 14, 15);
}

// ---------------------------------------------------------------------------
// f32 -> f16 cast
// ---------------------------------------------------------------------------
__global__ void cast_f32_f16(const float* __restrict__ s, _Float16* __restrict__ d, int n) {
    int i = blockIdx.x * blockDim.x + threadIdx.x;
    if (i < n) d[i] = (_Float16)s[i];
}

// ---------------------------------------------------------------------------
// RoPE on interleaved pairs within 64-dim sub-heads; f32 in, f16 out.
// ---------------------------------------------------------------------------
__global__ void rope_f32_f16(const float* __restrict__ s, _Float16* __restrict__ d,
                             const float* __restrict__ cosT, const float* __restrict__ sinT,
                             int width, int total) {
    int pid = blockIdx.x * blockDim.x + threadIdx.x;
    if (pid >= total) return;
    int ppr = width >> 1;
    int row = pid / ppr;
    int p   = pid - row * ppr;
    int sub = p >> 5;
    int i   = p & 31;
    float c  = cosT[row * 32 + i];
    float sn = sinT[row * 32 + i];
    int base = row * width + sub * 64 + 2 * i;
    float tr = s[base], ti = s[base + 1];
    d[base]     = (_Float16)(tr * c - ti * sn);
    d[base + 1] = (_Float16)(tr * sn + ti * c);
}

// ---------------------------------------------------------------------------
// Tiled GEMM: C(MxN,f32) = A(MxK,f16,rm) * B(KxN,f16,rm)
// 128 threads = 4 waves; block tile 64x64; wave tile 32x32 (2x2 wmma frags).
// A staged row-major (async DMA); B staged TRANSPOSED so B-fragment loads are
// contiguous b128 reads.
// ---------------------------------------------------------------------------
__global__ __launch_bounds__(128) void gemm_f16f32(
    const _Float16* __restrict__ A, const _Float16* __restrict__ B,
    float* __restrict__ C, int M, int N, int K) {
    __shared__ __align__(16) _Float16 As[64][40];    // row-major, 80B pitch
    __shared__ __align__(16) _Float16 BsT[64][40];   // BsT[n][k], 80B pitch

    int tid  = threadIdx.x;
    int lane = tid & 31, wave = tid >> 5;
    int wm = wave >> 1, wn = wave & 1;
    int r = lane & 15, H = lane >> 4;
    int bm = blockIdx.y * 64, bn = blockIdx.x * 64;

    v8f acc[2][2];
#pragma unroll
    for (int i = 0; i < 2; ++i)
#pragma unroll
        for (int j = 0; j < 2; ++j)
#pragma unroll
            for (int e = 0; e < 8; ++e) acc[i][j][e] = 0.0f;

    for (int k0 = 0; k0 < K; k0 += 32) {
        // A tile 64x32: 256 x 16B, async DMA straight into LDS.
#pragma unroll
        for (int t = 0; t < 2; ++t) {
            int idx  = tid + t * 128;
            int arow = idx >> 2, ac = (idx & 3) * 8;
            cp16_g2s(&As[arow][ac], &A[(size_t)(bm + arow) * K + k0 + ac]);
            if (k0 + 32 < K)
                __builtin_prefetch(&A[(size_t)(bm + arow) * K + k0 + 32 + ac], 0, 0);
        }
        // B tile 32x64: coalesced global b128 reads, transposed scatter to LDS.
#pragma unroll
        for (int t = 0; t < 2; ++t) {
            int idx  = tid + t * 128;
            int brow = idx >> 3, bc = (idx & 7) * 8;
            uint4 raw = *reinterpret_cast<const uint4*>(&B[(size_t)(k0 + brow) * N + bn + bc]);
            const _Float16* hp = reinterpret_cast<const _Float16*>(&raw);
#pragma unroll
            for (int e = 0; e < 8; ++e) BsT[bc + e][brow] = hp[e];
            if (k0 + 32 < K)
                __builtin_prefetch(&B[(size_t)(k0 + 32 + brow) * N + bn + bc], 0, 0);
        }
        wait_cp();
        __syncthreads();

        v16h af[2], bf[2];
#pragma unroll
        for (int i = 0; i < 2; ++i) {
            const _Float16* ap = &As[wm * 32 + i * 16 + r][0];
            af[i] = frag2(&ap[H * 8], &ap[16 + H * 8]);
            const _Float16* bp = &BsT[wn * 32 + i * 16 + r][0];
            bf[i] = frag2(&bp[H * 8], &bp[16 + H * 8]);
        }
#pragma unroll
        for (int i = 0; i < 2; ++i)
#pragma unroll
            for (int j = 0; j < 2; ++j)
                acc[i][j] = __builtin_amdgcn_wmma_f32_16x16x32_f16(
                    false, af[i], false, bf[j], (short)0, acc[i][j], false, false);
        __syncthreads();
    }

#pragma unroll
    for (int i = 0; i < 2; ++i)
#pragma unroll
        for (int j = 0; j < 2; ++j)
#pragma unroll
            for (int e = 0; e < 8; ++e) {
                int row = bm + wm * 32 + i * 16 + e + 8 * H;
                int col = bn + wn * 32 + j * 16 + r;
                C[(size_t)row * N + col] = acc[i][j][e];
            }
}

// ---------------------------------------------------------------------------
// Differential flash attention. One wave per (16-query tile, head).
// K tile staged row-major (fragment loads are naturally contiguous);
// V tile staged transposed with zero-padded k=16..31 so P(16x32,zero-padded) x
// V(32x16) fragments are contiguous b128 reads.
// ---------------------------------------------------------------------------
__global__ __launch_bounds__(32) void diff_attn(
    const _Float16* __restrict__ Qh, const _Float16* __restrict__ Kh,
    const _Float16* __restrict__ Vh, _Float16* __restrict__ Ah) {
    const float scale = 0.125f;   // 1/sqrt(64)
    const float LAM   = 0.2f;     // 0.8 - 0.6*exp(0)
    const float NEG   = -3.0e38f;

    int qt = blockIdx.x;
    int h  = blockIdx.y;
    int g  = h >> 1;
    int lane = threadIdx.x;
    int r = lane & 15, H = lane >> 4;

    __shared__ __align__(16) _Float16 Kt[16][136];   // 16 keys x 128 (K1|K2)
    __shared__ __align__(16) _Float16 VtT[128][40];  // VtT[n][k], k 0..31 (16..31 = 0)
    __shared__ __align__(16) _Float16 Pt[16][24];    // P tile 16x16

    // Zero the V padding rows once (multiplied by A-zeros; must be finite).
    for (int n = lane; n < 128; n += 32)
#pragma unroll
        for (int e = 0; e < 16; ++e) VtT[n][16 + e] = (_Float16)0.0f;

    // Q fragments (contiguous b128 global loads).
    const _Float16* Qbase = Qh + (size_t)(qt * 16) * 2048 + (size_t)h * 128 + (size_t)r * 2048;
    v16h q1a = frag2(&Qbase[H * 8],      &Qbase[16 + H * 8]);
    v16h q1b = frag2(&Qbase[32 + H * 8], &Qbase[48 + H * 8]);
    v16h q2a = frag2(&Qbase[64 + H * 8], &Qbase[80 + H * 8]);
    v16h q2b = frag2(&Qbase[96 + H * 8], &Qbase[112 + H * 8]);

    v8f o1[8], o2[8];
    float m1[8], l1[8], m2[8], l2[8];
#pragma unroll
    for (int c = 0; c < 8; ++c)
#pragma unroll
        for (int e = 0; e < 8; ++e) { o1[c][e] = 0.0f; o2[c][e] = 0.0f; }
#pragma unroll
    for (int e = 0; e < 8; ++e) { m1[e] = NEG; m2[e] = NEG; l1[e] = 0.0f; l2[e] = 0.0f; }

    for (int jt = 0; jt <= qt; ++jt) {
        // Stage K (async DMA, row-major) and V (transposed scatter).
#pragma unroll
        for (int t = 0; t < 8; ++t) {
            int idx = lane + t * 32;
            int row = idx >> 4, c8 = (idx & 15) * 8;
            size_t goff = (size_t)(jt * 16 + row) * 1024 + (size_t)g * 128 + c8;
            cp16_g2s(&Kt[row][c8], &Kh[goff]);
            uint4 raw = *reinterpret_cast<const uint4*>(&Vh[goff]);
            const _Float16* hp = reinterpret_cast<const _Float16*>(&raw);
#pragma unroll
            for (int e = 0; e < 8; ++e) VtT[c8 + e][row] = hp[e];
        }
        wait_cp();
        __syncthreads();

        // Scores: S1 = Q1*K1^T, S2 = Q2*K2^T  (B[k][n] = Kt[n][d0+k])
        v8f s1, s2;
#pragma unroll
        for (int e = 0; e < 8; ++e) { s1[e] = 0.0f; s2[e] = 0.0f; }
        {
            const _Float16* kp = &Kt[r][0];
            v16h kb;
            kb = frag2(&kp[H * 8],      &kp[16 + H * 8]);
            s1 = __builtin_amdgcn_wmma_f32_16x16x32_f16(false, q1a, false, kb, (short)0, s1, false, false);
            kb = frag2(&kp[32 + H * 8], &kp[48 + H * 8]);
            s1 = __builtin_amdgcn_wmma_f32_16x16x32_f16(false, q1b, false, kb, (short)0, s1, false, false);
            kb = frag2(&kp[64 + H * 8], &kp[80 + H * 8]);
            s2 = __builtin_amdgcn_wmma_f32_16x16x32_f16(false, q2a, false, kb, (short)0, s2, false, false);
            kb = frag2(&kp[96 + H * 8], &kp[112 + H * 8]);
            s2 = __builtin_amdgcn_wmma_f32_16x16x32_f16(false, q2b, false, kb, (short)0, s2, false, false);
        }

        // Online softmax update, both streams (C frag: row = e+8H, col = r).
        float p1[8], p2[8], a1[8], a2[8];
        bool diag = (jt == qt);
#pragma unroll
        for (int e = 0; e < 8; ++e) {
            float v1 = s1[e] * scale, v2 = s2[e] * scale;
            if (diag && (r > e + 8 * H)) { v1 = NEG; v2 = NEG; }
            float t1 = v1, t2 = v2;
#pragma unroll
            for (int ms = 1; ms < 16; ms <<= 1) {
                t1 = fmaxf(t1, __shfl_xor(t1, ms, 32));
                t2 = fmaxf(t2, __shfl_xor(t2, ms, 32));
            }
            float mn1 = fmaxf(m1[e], t1), mn2 = fmaxf(m2[e], t2);
            a1[e] = __expf(m1[e] - mn1);
            a2[e] = __expf(m2[e] - mn2);
            float e1 = __expf(v1 - mn1), e2 = __expf(v2 - mn2);
            float rs1 = e1, rs2 = e2;
#pragma unroll
            for (int ms = 1; ms < 16; ms <<= 1) {
                rs1 += __shfl_xor(rs1, ms, 32);
                rs2 += __shfl_xor(rs2, ms, 32);
            }
            l1[e] = l1[e] * a1[e] + rs1;
            l2[e] = l2[e] * a2[e] + rs2;
            m1[e] = mn1; m2[e] = mn2;
            p1[e] = e1;  p2[e] = e2;
        }
#pragma unroll
        for (int c = 0; c < 8; ++c)
#pragma unroll
            for (int e = 0; e < 8; ++e) { o1[c][e] *= a1[e]; o2[c][e] *= a2[e]; }

        // P1/P2 -> f16 A-fragments via LDS; valid halves are contiguous.
#pragma unroll
        for (int e = 0; e < 8; ++e) Pt[e + 8 * H][r] = (_Float16)p1[e];
        __syncthreads();
        v16h pa;
        {
            v8h lo = *reinterpret_cast<const v8h*>(&Pt[r][H * 8]);
            v8h zz;
#pragma unroll
            for (int e = 0; e < 8; ++e) zz[e] = (_Float16)0.0f;
            pa = __builtin_shufflevector(lo, zz, 0, 1, 2, 3, 4, 5, 6, 7, 8, 9, 10, 11, 12, 13, 14, 15);
        }
        __syncthreads();
#pragma unroll
        for (int e = 0; e < 8; ++e) Pt[e + 8 * H][r] = (_Float16)p2[e];
        __syncthreads();
        v16h pb;
        {
            v8h lo = *reinterpret_cast<const v8h*>(&Pt[r][H * 8]);
            v8h zz;
#pragma unroll
            for (int e = 0; e < 8; ++e) zz[e] = (_Float16)0.0f;
            pb = __builtin_shufflevector(lo, zz, 0, 1, 2, 3, 4, 5, 6, 7, 8, 9, 10, 11, 12, 13, 14, 15);
        }
        __syncthreads();

        // O += P * V over 8 column chunks; V fragments contiguous from VtT.
#pragma unroll
        for (int c = 0; c < 8; ++c) {
            const _Float16* vp = &VtT[c * 16 + r][0];
            v16h vb = frag2(&vp[H * 8], &vp[16 + H * 8]);
            o1[c] = __builtin_amdgcn_wmma_f32_16x16x32_f16(false, pa, false, vb, (short)0, o1[c], false, false);
            o2[c] = __builtin_amdgcn_wmma_f32_16x16x32_f16(false, pb, false, vb, (short)0, o2[c], false, false);
        }
        __syncthreads();
    }

    // Combine: out = O1/l1 - LAM * O2/l2, store f16.
    _Float16* Abase = Ah + (size_t)(qt * 16) * 2048 + (size_t)h * 128;
#pragma unroll
    for (int c = 0; c < 8; ++c)
#pragma unroll
        for (int e = 0; e < 8; ++e) {
            float val = o1[c][e] / l1[e] - LAM * (o2[c][e] / l2[e]);
            Abase[(size_t)(e + 8 * H) * 2048 + c * 16 + r] = (_Float16)val;
        }
}

// ---------------------------------------------------------------------------
// Host launcher
// ---------------------------------------------------------------------------
extern "C" void kernel_launch(void* const* d_in, const int* in_sizes, int n_in,
                              void* d_out, int out_size, void* d_ws, size_t ws_size,
                              hipStream_t stream) {
    const float* x    = (const float*)d_in[0];
    const float* cosT = (const float*)d_in[1];
    const float* sinT = (const float*)d_in[2];
    const float* Wq   = (const float*)d_in[3];
    const float* Wk   = (const float*)d_in[4];
    const float* Wv   = (const float*)d_in[5];
    const float* Wo   = (const float*)d_in[6];
    float* out = (float*)d_out;

    const size_t MB = 1024 * 1024;
    char* ws = (char*)d_ws;
    _Float16* xh  = (_Float16*)(ws + 0 * MB);
    _Float16* Wqh = (_Float16*)(ws + 8 * MB);
    _Float16* Wkh = (_Float16*)(ws + 16 * MB);
    _Float16* Wvh = (_Float16*)(ws + 20 * MB);
    _Float16* Woh = (_Float16*)(ws + 24 * MB);
    float*    Qf  = (float*)  (ws + 32 * MB);
    float*    Kf  = (float*)  (ws + 48 * MB);
    float*    Vf  = (float*)  (ws + 56 * MB);
    _Float16* Qhh = (_Float16*)(ws + 64 * MB);
    _Float16* Khh = (_Float16*)(ws + 72 * MB);
    _Float16* Vhh = (_Float16*)(ws + 76 * MB);
    _Float16* Ahh = (_Float16*)(ws + 80 * MB);

    const int L = 2048, D = 2048, DK = 1024;

    cast_f32_f16<<<(L * D + 255) / 256, 256, 0, stream>>>(x, xh, L * D);
    cast_f32_f16<<<(D * D + 255) / 256, 256, 0, stream>>>(Wq, Wqh, D * D);
    cast_f32_f16<<<(D * DK + 255) / 256, 256, 0, stream>>>(Wk, Wkh, D * DK);
    cast_f32_f16<<<(D * DK + 255) / 256, 256, 0, stream>>>(Wv, Wvh, D * DK);
    cast_f32_f16<<<(D * D + 255) / 256, 256, 0, stream>>>(Wo, Woh, D * D);

    dim3 blk(128);
    gemm_f16f32<<<dim3(D / 64, L / 64), blk, 0, stream>>>(xh, Wqh, Qf, L, D, D);
    gemm_f16f32<<<dim3(DK / 64, L / 64), blk, 0, stream>>>(xh, Wkh, Kf, L, DK, D);
    gemm_f16f32<<<dim3(DK / 64, L / 64), blk, 0, stream>>>(xh, Wvh, Vf, L, DK, D);

    int qpairs = L * (D / 2), kpairs = L * (DK / 2);
    rope_f32_f16<<<(qpairs + 255) / 256, 256, 0, stream>>>(Qf, Qhh, cosT, sinT, D, qpairs);
    rope_f32_f16<<<(kpairs + 255) / 256, 256, 0, stream>>>(Kf, Khh, cosT, sinT, DK, kpairs);
    cast_f32_f16<<<(L * DK + 255) / 256, 256, 0, stream>>>(Vf, Vhh, L * DK);

    diff_attn<<<dim3(L / 16, 16), 32, 0, stream>>>(Qhh, Khh, Vhh, Ahh);

    gemm_f16f32<<<dim3(D / 64, L / 64), blk, 0, stream>>>(Ahh, Woh, out, L, D, D);
}
